// HeteroGATEncoder_65472481460404
// MI455X (gfx1250) — matrix operand order repaired and checked by reading, not compiled, not run on hardware
//
#include <hip/hip_runtime.h>
#include <hip/hip_bf16.h>
#include <math.h>

typedef __bf16 bf16_t;
typedef __attribute__((ext_vector_type(16))) __bf16 v16bf;
typedef __attribute__((ext_vector_type(8)))  __bf16 v8bf;
typedef __attribute__((ext_vector_type(8)))  float  v8f;

// ---------------------------------------------------------------- helpers

__device__ __forceinline__ unsigned short bf_bits(float f) {
    unsigned u = __float_as_uint(f);
    u += 0x7fffu + ((u >> 16) & 1u);          // round-to-nearest-even
    return (unsigned short)(u >> 16);
}

__device__ __forceinline__ bf16_t f2bf(float f) {
    unsigned short s = bf_bits(f);
    bf16_t r;
    __builtin_memcpy(&r, &s, 2);
    return r;
}

__device__ __forceinline__ float atomicMaxF(float* addr, float val) {
    if (val >= 0.0f)
        return __int_as_float(atomicMax((int*)addr, __float_as_int(val)));
    else
        return __uint_as_float(atomicMin((unsigned int*)addr, __float_as_uint(val)));
}

// ---------------------------------------------------------------- kernels

// Pack fp32 W[K,N] -> bf16 Bp[(k/32)*N*32 + n*32 + (k%32)]  (WMMA-fragment-native)
__global__ void hgat_pack_w_bf16(const float* __restrict__ W,
                                 bf16_t* __restrict__ Bp, int K, int N) {
    int idx = blockIdx.x * blockDim.x + threadIdx.x;
    if (idx >= K * N) return;
    int kk = idx & 31;
    int n  = (idx >> 5) % N;
    int kt = idx / (32 * N);
    int k  = kt * 32 + kk;
    Bp[idx] = f2bf(W[(size_t)k * N + n]);
}

__global__ void hgat_fill_f32(float* __restrict__ p, float v, int n) {
    int i = blockIdx.x * blockDim.x + threadIdx.x;
    if (i < n) p[i] = v;
}

// C[M,N] = A[M,K](fp32 -> bf16) @ B[K,N](pre-packed bf16) (+ bias), fp32 acc via WMMA.
// Block: 256 threads = 8 waves. BM=128, BN=128, BK=32. Wave w owns rows [w*16, w*16+16).
__global__ __launch_bounds__(256) void hgat_gemm_bf16_wmma(
    const float* __restrict__ A, const bf16_t* __restrict__ Bp,
    float* __restrict__ C, const float* __restrict__ bias,
    int M, int N, int K) {
    __shared__ bf16_t As[128 * 32];   // 8 KB, row-major: As[m*32 + k]
    __shared__ bf16_t Bs[128 * 32];   // 8 KB, col-packed: Bs[n*32 + k]

    const int tid  = threadIdx.x;
    const int wave = tid >> 5;
    const int lane = tid & 31;
    const int hi   = lane >> 4;       // half-wave select
    const int l16  = lane & 15;
    const int m0   = blockIdx.x * 128;
    const int n0   = blockIdx.y * 128;

    v8f acc[8] = {};                  // eight 16x16 n-tiles per wave

    for (int k0 = 0; k0 < K; k0 += 32) {
        // Stage A tile: float4 loads, bf16 convert, ushort4 (8B) LDS stores.
        // 128 rows x 8 float4-chunks = 1024 chunks, 4 per thread.
        for (int ch = tid; ch < 128 * 8; ch += 256) {
            int r  = ch >> 3;
            int c4 = (ch & 7) * 4;
            int gr = m0 + r;
            float4 v = make_float4(0.0f, 0.0f, 0.0f, 0.0f);
            if (gr < M) v = *(const float4*)&A[(size_t)gr * K + k0 + c4];
            ushort4 o;
            o.x = bf_bits(v.x); o.y = bf_bits(v.y);
            o.z = bf_bits(v.z); o.w = bf_bits(v.w);
            *(ushort4*)&As[r * 32 + c4] = o;
        }
        // Stage B tile: contiguous copy of 4096 bf16 (8 KB) = 512 uint4, 2 per thread.
        {
            const uint4* srcv = (const uint4*)&Bp[((size_t)(k0 >> 5) * N + n0) * 32];
            uint4* dstv = (uint4*)Bs;
            for (int i = tid; i < 512; i += 256) dstv[i] = srcv[i];
        }
        __syncthreads();

        // A fragment: two contiguous 16B chunks -> 2x ds_load_b128
        const int arow = wave * 16 + l16;
        v8bf a0 = *(const v8bf*)&As[arow * 32 + hi * 8];
        v8bf a1 = *(const v8bf*)&As[arow * 32 + 16 + hi * 8];
        v16bf a = __builtin_shufflevector(a0, a1, 0, 1, 2, 3, 4, 5, 6, 7,
                                          8, 9, 10, 11, 12, 13, 14, 15);
#pragma unroll
        for (int t = 0; t < 8; ++t) {
            // B fragment: two contiguous 16B chunks (k = hi*8.., 16+hi*8..) -> 2x ds_load_b128
            const int bcol = t * 16 + l16;
            v8bf b0 = *(const v8bf*)&Bs[bcol * 32 + hi * 8];
            v8bf b1 = *(const v8bf*)&Bs[bcol * 32 + 16 + hi * 8];
            v16bf b = __builtin_shufflevector(b0, b1, 0, 1, 2, 3, 4, 5, 6, 7,
                                              8, 9, 10, 11, 12, 13, 14, 15);
            acc[t] = __builtin_amdgcn_wmma_f32_16x16x32_bf16(
                false, a, false, b, (short)0, acc[t], false, false);
        }
        __syncthreads();
    }

    // Epilogue: vgpr r -> row m0+wave*16+hi*8+r, lane l16 -> col n0+t*16+l16
#pragma unroll
    for (int t = 0; t < 8; ++t) {
#pragma unroll
        for (int r = 0; r < 8; ++r) {
            int gm = m0 + wave * 16 + hi * 8 + r;
            int gn = n0 + t * 16 + l16;
            if (gm < M) {
                float v = acc[t][r];
                if (bias) v += bias[gn];
                C[(size_t)gm * N + gn] = v;
            }
        }
    }
}

// w[k,h] = sum_c W[k, h*64+c] * att[h*64+c]   (K=256, H=4)
__global__ void hgat_make_watt(const float* __restrict__ W,
                               const float* __restrict__ att,
                               float* __restrict__ w) {
    int idx = blockIdx.x * blockDim.x + threadIdx.x;   // 1024 threads
    if (idx >= 256 * 4) return;
    int k = idx >> 2, h = idx & 3;
    float acc = 0.0f;
#pragma unroll 8
    for (int c = 0; c < 64; ++c)
        acc += W[(size_t)k * 256 + h * 64 + c] * att[h * 64 + c];
    w[idx] = acc;
}

// a[n,h] = sum_k h[n,k] * w[k,h]
__global__ void hgat_att_reduce(const float* __restrict__ hfeat,
                                const float* __restrict__ w,
                                float* __restrict__ out, int Nn) {
    int idx = blockIdx.x * blockDim.x + threadIdx.x;
    if (idx >= Nn * 4) return;
    int n = idx >> 2, h = idx & 3;
    float acc = 0.0f;
#pragma unroll 8
    for (int k = 0; k < 256; ++k)
        acc += hfeat[(size_t)n * 256 + k] * w[k * 4 + h];
    out[idx] = acc;
}

// acc row-init to sum of conv biases
__global__ void hgat_init_acc(float* __restrict__ acc,
                              const float* __restrict__ b0,
                              const float* __restrict__ b1,
                              const float* __restrict__ b2, int total) {
    int idx = blockIdx.x * blockDim.x + threadIdx.x;
    if (idx >= total) return;
    int c = idx & 255;
    float v = b0[c];
    if (b1) v += b1[c];
    if (b2) v += b2[c];
    acc[idx] = v;
}

// logit = leaky_relu(a_s[src]+a_d[dst], 0.2); atomicMax per (dst,h)
__global__ void hgat_edge_logits(const int* __restrict__ src,
                                 const int* __restrict__ dst, int E, int Etot,
                                 const float* __restrict__ a_s,
                                 const float* __restrict__ a_d,
                                 float* __restrict__ logits,
                                 float* __restrict__ mbuf) {
    int idx = blockIdx.x * blockDim.x + threadIdx.x;
    if (idx >= Etot * 4) return;
    int e = idx >> 2, h = idx & 3;
    int s = (e < E) ? src[e] : (e - E);
    int d = (e < E) ? dst[e] : (e - E);
    float x = a_s[s * 4 + h] + a_d[d * 4 + h];
    x = (x > 0.0f) ? x : 0.2f * x;
    logits[idx] = x;
    atomicMaxF(&mbuf[d * 4 + h], x);
}

// eu = exp(logit - m[dst]); atomicAdd into s[dst]; logits overwritten with eu
__global__ void hgat_edge_exp(const int* __restrict__ dst, int E, int Etot,
                              float* __restrict__ logits,
                              const float* __restrict__ mbuf,
                              float* __restrict__ sbuf) {
    int idx = blockIdx.x * blockDim.x + threadIdx.x;
    if (idx >= Etot * 4) return;
    int e = idx >> 2, h = idx & 3;
    int d = (e < E) ? dst[e] : (e - E);
    float eu = expf(logits[idx] - mbuf[d * 4 + h]);
    logits[idx] = eu;
    atomicAdd(&sbuf[d * 4 + h], eu);
}

// acc[dst, c] += hs[src, c] * eu[e, c/64] / (s[dst, c/64] + 1e-16)
__global__ void hgat_edge_scatter(const int* __restrict__ src,
                                  const int* __restrict__ dst, int E,
                                  const float* __restrict__ hs,
                                  const float* __restrict__ eu,
                                  const float* __restrict__ sbuf,
                                  float* __restrict__ acc) {
    int e = blockIdx.x;               // one block (256 threads) per edge
    int c = threadIdx.x;
    int h = c >> 6;
    int s = (e < E) ? src[e] : (e - E);
    int d = (e < E) ? dst[e] : (e - E);
    float alpha = eu[e * 4 + h] / (sbuf[d * 4 + h] + 1e-16f);
    atomicAdd(&acc[(size_t)d * 256 + c], hs[(size_t)s * 256 + c] * alpha);
}

// LayerNorm + ReLU (+ residual into h)
__global__ __launch_bounds__(256) void hgat_ln_relu(
    const float* __restrict__ acc, const float* __restrict__ g,
    const float* __restrict__ b, float* __restrict__ h, int residual) {
    __shared__ float sm[256];
    int row = blockIdx.x, tid = threadIdx.x;
    float x = acc[(size_t)row * 256 + tid];
    sm[tid] = x;
    __syncthreads();
    for (int s = 128; s > 0; s >>= 1) {
        if (tid < s) sm[tid] += sm[tid + s];
        __syncthreads();
    }
    float mu = sm[0] * (1.0f / 256.0f);
    __syncthreads();
    float dx = x - mu;
    sm[tid] = dx * dx;
    __syncthreads();
    for (int s = 128; s > 0; s >>= 1) {
        if (tid < s) sm[tid] += sm[tid + s];
        __syncthreads();
    }
    float var = sm[0] * (1.0f / 256.0f);
    float y = dx * rsqrtf(var + 1e-5f) * g[tid] + b[tid];
    y = fmaxf(y, 0.0f);
    if (residual) y += h[(size_t)row * 256 + tid];
    h[(size_t)row * 256 + tid] = y;
}

// ---------------------------------------------------------------- host

extern "C" void kernel_launch(void* const* d_in, const int* in_sizes, int n_in,
                              void* d_out, int out_size, void* d_ws, size_t ws_size,
                              hipStream_t stream) {
    (void)in_sizes; (void)n_in; (void)out_size; (void)ws_size;
    constexpr int NG = 20000, NC = 200000, NM = 2000, HIDc = 256;

    const float* x_gene  = (const float*)d_in[0];
    const float* x_cpg   = (const float*)d_in[1];
    const float* x_mirna = (const float*)d_in[2];

    struct Edge { const int* src; const int* dst; int E; };
    Edge e_maps       { (const int*)d_in[3],  (const int*)d_in[4],  200000 };
    Edge e_targets    { (const int*)d_in[5],  (const int*)d_in[6],  100000 };
    Edge e_ppi        { (const int*)d_in[7],  (const int*)d_in[8],  400000 };
    Edge e_rev_maps   { (const int*)d_in[9],  (const int*)d_in[10], 200000 };
    Edge e_rev_tgts   { (const int*)d_in[11], (const int*)d_in[12], 100000 };

    // params flattened via jax pytree (sorted dict keys):
    // layers: per layer, edge keys sorted [maps, ppi, rev_maps, rev_targets, targets],
    //         per edge fields sorted [W, att_dst, att_src, bias]
    auto LP = [&](int layer, int sidx, int field) -> const float* {
        return (const float*)d_in[13 + layer * 20 + sidx * 4 + field];
    };
    // ln: per layer, node keys sorted [cpg, gene, mirna], fields sorted [b, g]
    auto LNp = [&](int layer, int idx) -> const float* {
        return (const float*)d_in[53 + layer * 6 + idx];
    };
    const float* projW_cpg   = (const float*)d_in[65];
    const float* projB_cpg   = (const float*)d_in[66];
    const float* projW_gene  = (const float*)d_in[67];
    const float* projB_gene  = (const float*)d_in[68];
    const float* projW_mirna = (const float*)d_in[69];
    const float* projB_mirna = (const float*)d_in[70];

    // ---- workspace carve
    char* ws = (char*)d_ws;
    size_t off = 0;
    auto carve = [&](size_t bytes) -> void* {
        void* p = ws + off;
        off = (off + bytes + 255) & ~(size_t)255;
        return p;
    };
    float* acc_gene  = (float*)carve((size_t)NG * HIDc * 4);
    float* acc_cpg   = (float*)carve((size_t)NC * HIDc * 4);
    float* acc_mirna = (float*)carve((size_t)NM * HIDc * 4);
    float* S         = (float*)carve((size_t)NC * HIDc * 4);   // hs, max Ns
    float* logits    = (float*)carve((size_t)420000 * 4 * 4);
    float* mbuf      = (float*)carve((size_t)NC * 4 * 4);
    float* sbuf      = (float*)carve((size_t)NC * 4 * 4);
    float* a_s       = (float*)carve((size_t)NC * 4 * 4);
    float* a_d       = (float*)carve((size_t)NC * 4 * 4);
    float* w_s       = (float*)carve(256 * 4 * 4);
    float* w_d       = (float*)carve(256 * 4 * 4);
    bf16_t* wb_layer[2][5];
    for (int l = 0; l < 2; ++l)
        for (int c = 0; c < 5; ++c)
            wb_layer[l][c] = (bf16_t*)carve((size_t)65536 * 2);
    bf16_t* wb_pgene  = (bf16_t*)carve((size_t)65536 * 2);
    bf16_t* wb_pcpg   = (bf16_t*)carve((size_t)32768 * 2);
    bf16_t* wb_pmirna = (bf16_t*)carve((size_t)32768 * 2);

    auto packw = [&](const float* src, bf16_t* dst, int K) {
        int n = K * HIDc;
        hgat_pack_w_bf16<<<dim3((n + 255) / 256), dim3(256), 0, stream>>>(src, dst, K, HIDc);
    };
    auto gemm = [&](const float* A, const bf16_t* Bp, float* C,
                    const float* bias, int M, int K) {
        dim3 grid((M + 127) / 128, HIDc / 128);
        hgat_gemm_bf16_wmma<<<grid, dim3(256), 0, stream>>>(A, Bp, C, bias, M, HIDc, K);
    };

    // h lives in d_out: [gene | cpg | mirna]
    float* h_gene  = (float*)d_out;
    float* h_cpg   = h_gene + (size_t)NG * HIDc;
    float* h_mirna = h_cpg + (size_t)NC * HIDc;

    // ---- weight packing (once per launch)
    packw(projW_gene,  wb_pgene,  256);
    packw(projW_cpg,   wb_pcpg,   128);
    packw(projW_mirna, wb_pmirna, 128);
    for (int l = 0; l < 2; ++l)
        for (int c = 0; c < 5; ++c)
            packw(LP(l, c, 0), wb_layer[l][c], 256);

    // ---- projections
    gemm(x_gene,  wb_pgene,  h_gene,  projB_gene,  NG, 256);
    gemm(x_cpg,   wb_pcpg,   h_cpg,   projB_cpg,   NC, 128);
    gemm(x_mirna, wb_pmirna, h_mirna, projB_mirna, NM, 128);

    // ---- one GAT conv: fields W=0, att_dst=1, att_src=2, bias=3
    auto run_conv = [&](const float* hsrc, int Ns, const float* hdst, int Nd,
                        Edge e, int nloop, const bf16_t* Wb, const float* W,
                        const float* att_src, const float* att_dst, float* acc) {
        hgat_make_watt<<<dim3(4), dim3(256), 0, stream>>>(W, att_src, w_s);
        hgat_make_watt<<<dim3(4), dim3(256), 0, stream>>>(W, att_dst, w_d);
        hgat_att_reduce<<<dim3((Ns * 4 + 255) / 256), dim3(256), 0, stream>>>(hsrc, w_s, a_s, Ns);
        hgat_att_reduce<<<dim3((Nd * 4 + 255) / 256), dim3(256), 0, stream>>>(hdst, w_d, a_d, Nd);
        gemm(hsrc, Wb, S, nullptr, Ns, 256);                       // hs (messages)
        int Etot = e.E + nloop;
        hgat_fill_f32<<<dim3((Nd * 4 + 255) / 256), dim3(256), 0, stream>>>(mbuf, -INFINITY, Nd * 4);
        hgat_fill_f32<<<dim3((Nd * 4 + 255) / 256), dim3(256), 0, stream>>>(sbuf, 0.0f, Nd * 4);
        hgat_edge_logits<<<dim3((Etot * 4 + 255) / 256), dim3(256), 0, stream>>>(
            e.src, e.dst, e.E, Etot, a_s, a_d, logits, mbuf);
        hgat_edge_exp<<<dim3((Etot * 4 + 255) / 256), dim3(256), 0, stream>>>(
            e.dst, e.E, Etot, logits, mbuf, sbuf);
        hgat_edge_scatter<<<dim3(Etot), dim3(256), 0, stream>>>(
            e.src, e.dst, e.E, S, logits, sbuf, acc);
    };

    for (int l = 0; l < 2; ++l) {
        // gene acc gets maps + targets + ppi biases; cpg gets rev_maps; mirna gets rev_targets
        hgat_init_acc<<<dim3((NG * 256 + 255) / 256), dim3(256), 0, stream>>>(
            acc_gene, LP(l, 0, 3), LP(l, 4, 3), LP(l, 1, 3), NG * 256);
        hgat_init_acc<<<dim3((NC * 256 + 255) / 256), dim3(256), 0, stream>>>(
            acc_cpg, LP(l, 2, 3), nullptr, nullptr, NC * 256);
        hgat_init_acc<<<dim3((NM * 256 + 255) / 256), dim3(256), 0, stream>>>(
            acc_mirna, LP(l, 3, 3), nullptr, nullptr, NM * 256);

        // maps (sorted idx 0): cpg -> gene
        run_conv(h_cpg, NC, h_gene, NG, e_maps, 0,
                 wb_layer[l][0], LP(l, 0, 0), LP(l, 0, 2), LP(l, 0, 1), acc_gene);
        // targets (sorted idx 4): mirna -> gene
        run_conv(h_mirna, NM, h_gene, NG, e_targets, 0,
                 wb_layer[l][4], LP(l, 4, 0), LP(l, 4, 2), LP(l, 4, 1), acc_gene);
        // ppi (sorted idx 1): gene -> gene with NG virtual self-loops
        run_conv(h_gene, NG, h_gene, NG, e_ppi, NG,
                 wb_layer[l][1], LP(l, 1, 0), LP(l, 1, 2), LP(l, 1, 1), acc_gene);
        // rev_maps (sorted idx 2): gene -> cpg
        run_conv(h_gene, NG, h_cpg, NC, e_rev_maps, 0,
                 wb_layer[l][2], LP(l, 2, 0), LP(l, 2, 2), LP(l, 2, 1), acc_cpg);
        // rev_targets (sorted idx 3): gene -> mirna
        run_conv(h_gene, NG, h_mirna, NM, e_rev_tgts, 0,
                 wb_layer[l][3], LP(l, 3, 0), LP(l, 3, 2), LP(l, 3, 1), acc_mirna);

        // ln order: cpg.b(0) cpg.g(1) gene.b(2) gene.g(3) mirna.b(4) mirna.g(5)
        hgat_ln_relu<<<dim3(NG), dim3(256), 0, stream>>>(acc_gene,  LNp(l, 3), LNp(l, 2), h_gene,  l);
        hgat_ln_relu<<<dim3(NC), dim3(256), 0, stream>>>(acc_cpg,   LNp(l, 1), LNp(l, 0), h_cpg,   l);
        hgat_ln_relu<<<dim3(NM), dim3(256), 0, stream>>>(acc_mirna, LNp(l, 5), LNp(l, 4), h_mirna, l);
    }
}
